// NACCEmbedder_60705067761778
// MI455X (gfx1250) — matrix operand order repaired and checked by reading, not compiled, not run on hardware
//
#include <hip/hip_runtime.h>
#include <hip/hip_bf16.h>
#include <math.h>

// ---------------- model constants ----------------
constexpr int Bsz = 256;      // batch
constexpr int Lseq = 256;     // sequence length
constexpr int Dm = 128;       // model dim
constexpr int FFd = 2048;     // ffn dim
constexpr int NL = 3;         // layers
constexpr int NH = 4;         // heads
constexpr int HD = 32;        // head dim
constexpr int Cc = 3;         // classes
constexpr int TD = 3 * Dm;    // 384
constexpr int BL = Bsz * Lseq; // 65536
constexpr float TAU_INV = 20.0f;   // 1/0.05
constexpr float LN_EPS = 1e-5f;

typedef _Float16 v16h __attribute__((ext_vector_type(16)));
typedef _Float16 v8h  __attribute__((ext_vector_type(8)));
typedef float    v8f  __attribute__((ext_vector_type(8)));
typedef int      v4i  __attribute__((__vector_size__(16)));

#define DEV static __device__ __forceinline__

// gfx1250 async global->LDS copy path (guarded: falls back to plain copies)
#if defined(__gfx1250__) && __has_builtin(__builtin_amdgcn_global_load_async_to_lds_b128)
#define HAVE_ASYNC_LDS 1
#else
#define HAVE_ASYNC_LDS 0
#endif

#if HAVE_ASYNC_LDS
DEV void async_copy_b128(const void* g, void* l) {
  // signature (from clang diagnostic): (v4i AS1*, v4i AS3*, imm offset, imm cpol)
  __builtin_amdgcn_global_load_async_to_lds_b128(
      (__attribute__((address_space(1))) v4i*)(void*)g,
      (__attribute__((address_space(3))) v4i*)l, 0, 0);
}
DEV void async_wait0() {
#if __has_builtin(__builtin_amdgcn_s_wait_asynccnt)
  __builtin_amdgcn_s_wait_asynccnt(0);
#else
  asm volatile("s_wait_asynccnt 0x0" ::: "memory");
#endif
}
#endif

DEV v8f wmma_f16f32(v16h a, v16h b, v8f c) {
  // D(16x16,f32) = A(16x32,f16) * B(32x16,f16) + C
  return __builtin_amdgcn_wmma_f32_16x16x32_f16(false, a, false, b, (short)0, c,
                                                false, false);
}

DEV v8f zero8() { v8f z = {0.f, 0.f, 0.f, 0.f, 0.f, 0.f, 0.f, 0.f}; return z; }

// A fragment: 16x32 f16, source row-major [16][stride] (stride in halfs, mult of 8).
// lane l (0..31): row m = l&15 ; kbase = (l<16)?0:8
// halfs 0..7  = src[m][kbase .. kbase+7]      (VGPR 0..3: K = kbase+2g+h)
// halfs 8..15 = src[m][kbase+16 .. kbase+23]  (VGPR 4..7: K = 16+kbase+...)
DEV v16h load_a_frag(const _Float16* row0, int stride, int lane) {
  int m = lane & 15;
  int kb = (lane < 16) ? 0 : 8;
  const _Float16* p = row0 + (size_t)m * stride + kb;
  v8h lo = *(const v8h*)(p);
  v8h hi = *(const v8h*)(p + 16);
  v16h a;
#pragma unroll
  for (int i = 0; i < 8; ++i) { a[i] = lo[i]; a[i + 8] = hi[i]; }
  return a;
}

// B fragment: 32x16 f16 with B[k][n] = W[n][k]; W row-major [16][stride].
// lane l: n = l&15 ; kbase = (l<16)?0:8 ; halfs 0..7 = W[n][kbase..kbase+7], 8..15 = W[n][kbase+16..kbase+23]
DEV v16h load_b_frag(const _Float16* row0, int stride, int lane) {
  int n = lane & 15;
  int ks = (lane < 16) ? 0 : 8;
  const _Float16* p = row0 + (size_t)n * stride + ks;
  v8h lo = *(const v8h*)(p);
  v8h hi = *(const v8h*)(p + 16);
  v16h b;
#pragma unroll
  for (int i = 0; i < 8; ++i) { b[i] = lo[i]; b[i + 8] = hi[i]; }
  return b;
}
// C fragment element r: row = r + ((lane<16)?0:8), col = lane&15.

// ---------------- elementwise helpers ----------------
__global__ __launch_bounds__(256) void cvt_f32_to_f16_kernel(const float* __restrict__ src,
                                                             _Float16* __restrict__ dst, int n) {
  int i = blockIdx.x * 256 + threadIdx.x;
  if (i < n) dst[i] = (_Float16)src[i];
}

// base[b,l,d] = x[b,l]*w0[d] + b0[d]
__global__ __launch_bounds__(256) void embed_kernel(const float* __restrict__ x,
                                                    const float* __restrict__ w0,
                                                    const float* __restrict__ b0,
                                                    float* __restrict__ zf,
                                                    _Float16* __restrict__ zh) {
  size_t idx = (size_t)blockIdx.x * 256 + threadIdx.x;
  if (idx >= (size_t)BL * Dm) return;
  int d = (int)(idx & (Dm - 1));
  size_t row = idx >> 7;  // /128
  float v = x[row] * w0[d] + b0[d];
  zf[idx] = v;
  zh[idx] = (_Float16)v;
}

// ---------------- generic WMMA GEMM: C = act(A @ W^T + bias) ----------------
// A: [M,K] f16 row-major ; W: [N,K] f16 row-major.
// Block tile 128(M) x 64(N); 8 waves, each wave owns a 32x32 tile (2x2 WMMA tiles)
// -> 4 fragment loads feed 4 v_wmma per K-step (2x reuse on A and B).
// grid (N/64, M/128), block 256.  act: 0=none 1=relu 2=tanh
__global__ __launch_bounds__(256) void gemm_kernel(const _Float16* __restrict__ A,
                                                   const _Float16* __restrict__ W,
                                                   const float* __restrict__ bias,
                                                   float* __restrict__ Cf,
                                                   _Float16* __restrict__ Ch,
                                                   int M, int N, int K, int act) {
  int lane = threadIdx.x & 31;
  int w = threadIdx.x >> 5;          // 8 waves: 4(m) x 2(n)
  int m0 = blockIdx.y * 128 + (w >> 1) * 32;
  int n0 = blockIdx.x * 64 + (w & 1) * 32;
  v8f acc[4] = {zero8(), zero8(), zero8(), zero8()};  // [mi*2+ni]
  const _Float16* ar0 = A + (size_t)m0 * K;
  const _Float16* ar1 = A + (size_t)(m0 + 16) * K;
  const _Float16* wr0 = W + (size_t)n0 * K;
  const _Float16* wr1 = W + (size_t)(n0 + 16) * K;
  for (int k0 = 0; k0 < K; k0 += 32) {
    __builtin_prefetch(ar0 + k0 + 32, 0, 1);   // global_prefetch_b8
    __builtin_prefetch(ar1 + k0 + 32, 0, 1);
    v16h a0 = load_a_frag(ar0 + k0, K, lane);
    v16h a1 = load_a_frag(ar1 + k0, K, lane);
    v16h b0 = load_b_frag(wr0 + k0, K, lane);
    v16h b1 = load_b_frag(wr1 + k0, K, lane);
    acc[0] = wmma_f16f32(a0, b0, acc[0]);
    acc[1] = wmma_f16f32(a0, b1, acc[1]);
    acc[2] = wmma_f16f32(a1, b0, acc[2]);
    acc[3] = wmma_f16f32(a1, b1, acc[3]);
  }
  int n15 = lane & 15;
  int mb = m0 + ((lane < 16) ? 0 : 8);
#pragma unroll
  for (int t = 0; t < 4; ++t) {
    int mrow = mb + (t >> 1) * 16;
    int ncol = n0 + (t & 1) * 16 + n15;
    float bv = bias ? bias[ncol] : 0.f;
#pragma unroll
    for (int r = 0; r < 8; ++r) {
      float v = acc[t][r] + bv;
      if (act == 1) v = fmaxf(v, 0.f);
      else if (act == 2) v = tanhf(v);
      size_t idx = (size_t)(mrow + r) * N + ncol;
      if (Cf) Cf[idx] = v;
      if (Ch) Ch[idx] = (_Float16)v;
    }
  }
}

// ---------------- flash attention (per b,h,128 q rows) ----------------
// qkvh: [B,L,3D] f16 (q|k|v). Writes Oh: [B,L,D] f16.
// grid (L/128, NH, Bsz), block 256 (8 waves x 16 q-rows).
__global__ __launch_bounds__(256) void attn_kernel(const _Float16* __restrict__ qkvh,
                                                   const unsigned char* __restrict__ mask,
                                                   _Float16* __restrict__ Oh) {
  __shared__ _Float16 Vt[HD][264];          // V^T, padded rows (528B = 16B-mult)
  __shared__ _Float16 Pl[8][16][40];        // per-wave P tile 16x32, stride 40
  __shared__ unsigned char msk[Lseq];
  int lane = threadIdx.x & 31;
  int w = threadIdx.x >> 5;
  int b = blockIdx.z, h = blockIdx.y;
  int qbase = blockIdx.x * 128;

  { // stage V^T (transpose) and mask
    int t = threadIdx.x;
    int n = t & (HD - 1);
    int ks = (t >> 5) * 32;
    const _Float16* vsrc = qkvh + (size_t)(b * Lseq + ks) * TD + 2 * Dm + h * HD + n;
#pragma unroll 8
    for (int i = 0; i < 32; ++i) Vt[n][ks + i] = vsrc[(size_t)i * TD];
    msk[t] = mask[(size_t)b * Lseq + t];
  }
  __syncthreads();

  int q0 = qbase + w * 16;
  const _Float16* qrow = qkvh + (size_t)(b * Lseq + q0) * TD + h * HD;
  v16h aq = load_a_frag(qrow, TD, lane);     // Q tile, reused across k

  float rowmax[8], rowsum[8];
#pragma unroll
  for (int r = 0; r < 8; ++r) { rowmax[r] = -1e30f; rowsum[r] = 0.f; }
  v8f o0 = zero8(), o1 = zero8();
  const float scale = 0.17677669529663687f;  // 1/sqrt(32)
  int n15 = lane & 15;

  for (int kt = 0; kt < Lseq; kt += 32) {
    const _Float16* krow = qkvh + (size_t)(b * Lseq + kt) * TD + Dm + h * HD;
    v16h bk0 = load_b_frag(krow, TD, lane);
    v16h bk1 = load_b_frag(krow + (size_t)16 * TD, TD, lane);
    v8f s0 = wmma_f16f32(aq, bk0, zero8());
    v8f s1 = wmma_f16f32(aq, bk1, zero8());
    bool m0f = msk[kt + n15] != 0;
    bool m1f = msk[kt + 16 + n15] != 0;
#pragma unroll
    for (int r = 0; r < 8; ++r) {
      float a0 = m0f ? -1e9f : s0[r] * scale;
      float a1 = m1f ? -1e9f : s1[r] * scale;
      float tmax = fmaxf(a0, a1);
      tmax = fmaxf(tmax, __shfl_xor(tmax, 1, 32));
      tmax = fmaxf(tmax, __shfl_xor(tmax, 2, 32));
      tmax = fmaxf(tmax, __shfl_xor(tmax, 4, 32));
      tmax = fmaxf(tmax, __shfl_xor(tmax, 8, 32));
      float nm = fmaxf(rowmax[r], tmax);
      float corr = expf(rowmax[r] - nm);
      rowmax[r] = nm;
      float p0 = expf(a0 - nm), p1 = expf(a1 - nm);
      float ts = p0 + p1;
      ts += __shfl_xor(ts, 1, 32);
      ts += __shfl_xor(ts, 2, 32);
      ts += __shfl_xor(ts, 4, 32);
      ts += __shfl_xor(ts, 8, 32);
      rowsum[r] = rowsum[r] * corr + ts;
      o0[r] *= corr;
      o1[r] *= corr;
      int mrow = r + ((lane < 16) ? 0 : 8);
      Pl[w][mrow][n15] = (_Float16)p0;        // wave-private: dscnt ordering suffices
      Pl[w][mrow][n15 + 16] = (_Float16)p1;
    }
    v16h ap  = load_a_frag(&Pl[w][0][0], 40, lane);
    v16h bv0 = load_b_frag(&Vt[0][kt], 264, lane);
    v16h bv1 = load_b_frag(&Vt[16][kt], 264, lane);
    o0 = wmma_f16f32(ap, bv0, o0);
    o1 = wmma_f16f32(ap, bv1, o1);
  }

  int mb = q0 + ((lane < 16) ? 0 : 8);
#pragma unroll
  for (int r = 0; r < 8; ++r) {
    float inv = 1.0f / rowsum[r];
    size_t base = (size_t)(b * Lseq + mb + r) * Dm + h * HD;
    Oh[base + n15] = (_Float16)(o0[r] * inv);
    Oh[base + 16 + n15] = (_Float16)(o1[r] * inv);
  }
}

// ---------------- fused FFN: G = relu(Z @ W1^T + b1) @ W2^T + b2 ----------------
// Zh: [M,128] f16 ; W1: [2048,128] f16 ; W2: [128,2048] f16 ; grid M/64, block 256.
// The 64x128 activation tile is staged once into LDS (async global->LDS on gfx1250).
__global__ __launch_bounds__(256) void ffn_kernel(const _Float16* __restrict__ Zh,
                                                  const _Float16* __restrict__ W1,
                                                  const float* __restrict__ B1,
                                                  const _Float16* __restrict__ W2,
                                                  const float* __restrict__ B2,
                                                  float* __restrict__ Gout) {
  __shared__ _Float16 zlds[64][Dm];           // 16 KB activation tile
  __shared__ _Float16 hlds[64][40];           // 64x32 hidden chunk, stride 40
  int lane = threadIdx.x & 31;
  int w = threadIdx.x >> 5;
  int row0 = blockIdx.x * 64;
  int mh = w & 3, nh = w >> 2;                // hidden tile owned by this wave
  int mo = w & 3, nog = (w >> 2) * 4;         // 4 output tiles per wave
  int n15 = lane & 15;

  { // stage Z tile: 8192 halfs, 16B per thread per iter
    int t = threadIdx.x;
    const _Float16* gsrc = Zh + (size_t)row0 * Dm + t * 8;
    _Float16* ldst = &zlds[0][0] + t * 8;
#if HAVE_ASYNC_LDS
#pragma unroll
    for (int i = 0; i < 4; ++i)
      async_copy_b128(gsrc + i * 2048, ldst + i * 2048);
    async_wait0();
#else
#pragma unroll
    for (int i = 0; i < 4; ++i)
      *(v8h*)(ldst + i * 2048) = *(const v8h*)(gsrc + i * 2048);
#endif
  }
  __syncthreads();

  v8f oacc[4] = {zero8(), zero8(), zero8(), zero8()};

  for (int fc = 0; fc < FFd; fc += 32) {
    // stage 1: hidden chunk h[64][32] = relu(z @ W1[fc:fc+32]^T + b1)
    v8f hacc = zero8();
    const _Float16* w1row = W1 + (size_t)(fc + nh * 16) * Dm;
#pragma unroll
    for (int k0 = 0; k0 < Dm; k0 += 32) {
      v16h a = load_a_frag(&zlds[mh * 16][k0], Dm, lane);
      v16h b = load_b_frag(w1row + k0, Dm, lane);
      hacc = wmma_f16f32(a, b, hacc);
    }
    float bv = B1[fc + nh * 16 + n15];
    int rb = mh * 16 + ((lane < 16) ? 0 : 8);
#pragma unroll
    for (int r = 0; r < 8; ++r)
      hlds[rb + r][nh * 16 + n15] = (_Float16)fmaxf(hacc[r] + bv, 0.f);
    __syncthreads();
    // stage 2: out += h @ W2[:, fc:fc+32]^T
    v16h ap = load_a_frag(&hlds[mo * 16][0], 40, lane);
#pragma unroll
    for (int j = 0; j < 4; ++j) {
      v16h b = load_b_frag(W2 + (size_t)((nog + j) * 16) * FFd + fc, FFd, lane);
      oacc[j] = wmma_f16f32(ap, b, oacc[j]);
    }
    __syncthreads();
  }
  int mb = row0 + mo * 16 + ((lane < 16) ? 0 : 8);
#pragma unroll
  for (int j = 0; j < 4; ++j) {
    int nc = (nog + j) * 16 + n15;
    float bv = B2[nc];
#pragma unroll
    for (int r = 0; r < 8; ++r)
      Gout[(size_t)(mb + r) * Dm + nc] = oacc[j][r] + bv;
  }
}

// ---------------- residual + layernorm (D=128, one wave per row) ----------------
__global__ __launch_bounds__(256) void resln_kernel(const float* __restrict__ Zin,
                                                    const float* __restrict__ G,
                                                    const float* __restrict__ S,
                                                    const float* __restrict__ Bb,
                                                    float* __restrict__ Zout,
                                                    _Float16* __restrict__ Zh) {
  int lane = threadIdx.x & 31;
  int w = threadIdx.x >> 5;
  size_t row = (size_t)blockIdx.x * 8 + w;
  const float* z = Zin + row * Dm;
  const float* g = G + row * Dm;
  float t[4];
  float s = 0.f;
#pragma unroll
  for (int i = 0; i < 4; ++i) { t[i] = z[lane + 32 * i] + g[lane + 32 * i]; s += t[i]; }
#pragma unroll
  for (int off = 1; off < 32; off <<= 1) s += __shfl_xor(s, off, 32);
  float mean = s * (1.0f / Dm);
  float vs = 0.f;
#pragma unroll
  for (int i = 0; i < 4; ++i) { float d = t[i] - mean; vs += d * d; }
#pragma unroll
  for (int off = 1; off < 32; off <<= 1) vs += __shfl_xor(vs, off, 32);
  float rstd = rsqrtf(vs * (1.0f / Dm) + LN_EPS);
#pragma unroll
  for (int i = 0; i < 4; ++i) {
    int d = lane + 32 * i;
    float v = (t[i] - mean) * rstd * S[d] + Bb[d];
    Zout[row * Dm + d] = v;
    Zh[row * Dm + d] = (_Float16)v;
  }
}

// ---------------- mean over sequence ----------------
__global__ __launch_bounds__(128) void seqmean_kernel(const float* __restrict__ Z,
                                                      float* __restrict__ Hm,
                                                      _Float16* __restrict__ Hmh) {
  int b = blockIdx.x, d = threadIdx.x;
  float s = 0.f;
  for (int l = 0; l < Lseq; ++l) s += Z[(size_t)(b * Lseq + l) * Dm + d];
  s *= (1.0f / Lseq);
  Hm[(size_t)b * Dm + d] = s;
  Hmh[(size_t)b * Dm + d] = (_Float16)s;
}

// ---------------- L2 normalize rows (one wave per row) ----------------
__global__ __launch_bounds__(256) void l2norm_kernel(const float* __restrict__ Hin,
                                                     float* __restrict__ Nout) {
  int lane = threadIdx.x & 31;
  int w = threadIdx.x >> 5;
  size_t row = (size_t)blockIdx.x * 8 + w;
  float t[4];
  float s = 0.f;
#pragma unroll
  for (int i = 0; i < 4; ++i) { t[i] = Hin[row * Dm + lane + 32 * i]; s += t[i] * t[i]; }
#pragma unroll
  for (int off = 1; off < 32; off <<= 1) s += __shfl_xor(s, off, 32);
  float inv = 1.0f / sqrtf(s);
#pragma unroll
  for (int i = 0; i < 4; ++i) Nout[row * Dm + lane + 32 * i] = t[i] * inv;
}

// ---------------- logits = softmax(hb @ w3^T + b3), C=3 ----------------
__global__ __launch_bounds__(256) void logits_kernel(const float* __restrict__ Hb,
                                                     const float* __restrict__ W3,
                                                     const float* __restrict__ B3,
                                                     float* __restrict__ Out) {
  int b = threadIdx.x;  // 256 threads, grid 1
  float acc[Cc];
#pragma unroll
  for (int c = 0; c < Cc; ++c) {
    float s = B3[c];
    for (int d = 0; d < Dm; ++d) s += Hb[(size_t)b * Dm + d] * W3[c * Dm + d];
    acc[c] = s;
  }
  float m = fmaxf(acc[0], fmaxf(acc[1], acc[2]));
  float e0 = expf(acc[0] - m), e1 = expf(acc[1] - m), e2 = expf(acc[2] - m);
  float inv = 1.0f / (e0 + e1 + e2);
  Out[b * Cc + 0] = e0 * inv;
  Out[b * Cc + 1] = e1 * inv;
  Out[b * Cc + 2] = e2 * inv;
}

// ---------------- InfoNCE per-row logsumexp - pos_sim ----------------
__global__ __launch_bounds__(128) void lse_kernel(const float* __restrict__ NB,
                                                  const float* __restrict__ NP,
                                                  const float* __restrict__ NN,
                                                  float* __restrict__ RowLoss) {
  __shared__ float nbq[Dm];
  __shared__ float red[128];
  int i = blockIdx.x, t = threadIdx.x;
  nbq[t] = NB[(size_t)i * Dm + t];
  __syncthreads();
  float dots[4];
#pragma unroll
  for (int jj = 0; jj < 4; ++jj) {
    int j = t + jj * 128;  // 0..511
    const float* src = (j < Bsz) ? NP + (size_t)j * Dm : NN + (size_t)(j - Bsz) * Dm;
    float s = 0.f;
    for (int d = 0; d < Dm; ++d) s += nbq[d] * src[d];
    dots[jj] = s * TAU_INV;
  }
  float lmax = fmaxf(fmaxf(dots[0], dots[1]), fmaxf(dots[2], dots[3]));
  red[t] = lmax;
  __syncthreads();
  for (int s = 64; s > 0; s >>= 1) { if (t < s) red[t] = fmaxf(red[t], red[t + s]); __syncthreads(); }
  float gmax = red[0];
  __syncthreads();
  float lsum = expf(dots[0] - gmax) + expf(dots[1] - gmax) +
               expf(dots[2] - gmax) + expf(dots[3] - gmax);
  red[t] = lsum;
  __syncthreads();
  for (int s = 64; s > 0; s >>= 1) { if (t < s) red[t] += red[t + s]; __syncthreads(); }
  float lse = gmax + logf(red[0]);
  __syncthreads();
  red[t] = nbq[t] * NP[(size_t)i * Dm + t];   // diag pos similarity
  __syncthreads();
  for (int s = 64; s > 0; s >>= 1) { if (t < s) red[t] += red[t + s]; __syncthreads(); }
  if (t == 0) RowLoss[i] = lse - red[0] * TAU_INV;
}

__global__ __launch_bounds__(256) void meanloss_kernel(const float* __restrict__ RowLoss,
                                                       float* __restrict__ out) {
  __shared__ float red[256];
  int t = threadIdx.x;
  red[t] = RowLoss[t];
  __syncthreads();
  for (int s = 128; s > 0; s >>= 1) { if (t < s) red[t] += red[t + s]; __syncthreads(); }
  if (t == 0) out[0] = red[0] * (1.0f / Bsz);
}

// ---------------- host orchestration ----------------
extern "C" void kernel_launch(void* const* d_in, const int* in_sizes, int n_in,
                              void* d_out, int out_size, void* d_ws, size_t ws_size,
                              hipStream_t stream) {
  (void)in_sizes; (void)n_in; (void)out_size; (void)ws_size;
  const float* xin[3] = {(const float*)d_in[0], (const float*)d_in[1], (const float*)d_in[2]};
  const unsigned char* mkin[3] = {(const unsigned char*)d_in[3], (const unsigned char*)d_in[4],
                                  (const unsigned char*)d_in[5]};
  const float* w0 = (const float*)d_in[6];
  const float* b0 = (const float*)d_in[7];
  const float* w_qkv = (const float*)d_in[8];
  const float* b_qkv = (const float*)d_in[9];
  const float* w_o = (const float*)d_in[10];
  const float* b_o = (const float*)d_in[11];
  const float* w_ff1 = (const float*)d_in[12];
  const float* b_ff1 = (const float*)d_in[13];
  const float* w_ff2 = (const float*)d_in[14];
  const float* b_ff2 = (const float*)d_in[15];
  const float* ln1_s = (const float*)d_in[16];
  const float* ln1_b = (const float*)d_in[17];
  const float* ln2_s = (const float*)d_in[18];
  const float* ln2_b = (const float*)d_in[19];
  const float* w1 = (const float*)d_in[20];
  const float* b1 = (const float*)d_in[21];
  const float* w2 = (const float*)d_in[22];
  const float* b2 = (const float*)d_in[23];
  const float* w3 = (const float*)d_in[24];
  const float* b3 = (const float*)d_in[25];
  float* out = (float*)d_out;
  constexpr int NB_OFF = 0, HP_OFF = 32768, HN_OFF = 65536, LG_OFF = 98304, LS_OFF = 99072;

  // ---- workspace carve (all 256B aligned; total ~156 MB) ----
  char* base = (char*)d_ws;
  size_t off = 0;
  auto alloc = [&](size_t bytes) -> void* {
    void* p = base + off;
    off = (off + bytes + 255) & ~(size_t)255;
    return p;
  };
  float*    zf     = (float*)   alloc((size_t)BL * Dm * 4);
  _Float16* zh     = (_Float16*)alloc((size_t)BL * Dm * 2);
  _Float16* qkvh   = (_Float16*)alloc((size_t)BL * TD * 2);
  _Float16* oh     = (_Float16*)alloc((size_t)BL * Dm * 2);
  float*    gf     = (float*)   alloc((size_t)BL * Dm * 4);
  _Float16* wqkv_h = (_Float16*)alloc((size_t)NL * TD * Dm * 2);
  _Float16* wo_h   = (_Float16*)alloc((size_t)NL * Dm * Dm * 2);
  _Float16* wff1_h = (_Float16*)alloc((size_t)NL * FFd * Dm * 2);
  _Float16* wff2_h = (_Float16*)alloc((size_t)NL * Dm * FFd * 2);
  _Float16* w1_h   = (_Float16*)alloc((size_t)Dm * Dm * 2);
  _Float16* w2_h   = (_Float16*)alloc((size_t)Dm * Dm * 2);
  float*    hm_f   = (float*)   alloc((size_t)Bsz * Dm * 4);
  _Float16* hm_h   = (_Float16*)alloc((size_t)Bsz * Dm * 2);
  float*    h1_f   = (float*)   alloc((size_t)Bsz * Dm * 4);
  _Float16* h1_h   = (_Float16*)alloc((size_t)Bsz * Dm * 2);
  float*    hb_f   = (float*)   alloc((size_t)Bsz * Dm * 4);
  float*    npn    = (float*)   alloc((size_t)Bsz * Dm * 4);
  float*    nnn    = (float*)   alloc((size_t)Bsz * Dm * 4);
  float*    rowls  = (float*)   alloc((size_t)Bsz * 4);

  auto cvt = [&](const float* s, _Float16* d, int n) {
    cvt_f32_to_f16_kernel<<<(n + 255) / 256, 256, 0, stream>>>(s, d, n);
  };
  cvt(w_qkv, wqkv_h, NL * TD * Dm);
  cvt(w_o, wo_h, NL * Dm * Dm);
  cvt(w_ff1, wff1_h, NL * FFd * Dm);
  cvt(w_ff2, wff2_h, NL * Dm * FFd);
  cvt(w1, w1_h, Dm * Dm);
  cvt(w2, w2_h, Dm * Dm);

  for (int br = 0; br < 3; ++br) {
    float* branch_out = (br == 0) ? hb_f : (br == 1 ? out + HP_OFF : out + HN_OFF);

    embed_kernel<<<(BL * Dm) / 256, 256, 0, stream>>>(xin[br], w0, b0, zf, zh);

    for (int l = 0; l < NL; ++l) {
      // QKV projection -> qkvh (f16 only)
      gemm_kernel<<<dim3(TD / 64, BL / 128), 256, 0, stream>>>(
          zh, wqkv_h + (size_t)l * TD * Dm, b_qkv + (size_t)l * TD,
          (float*)nullptr, qkvh, BL, TD, Dm, 0);
      // attention -> oh
      attn_kernel<<<dim3(Lseq / 128, NH, Bsz), 256, 0, stream>>>(qkvh, mkin[br], oh);
      // O projection -> gf (f32 only)
      gemm_kernel<<<dim3(Dm / 64, BL / 128), 256, 0, stream>>>(
          oh, wo_h + (size_t)l * Dm * Dm, b_o + (size_t)l * Dm,
          gf, (_Float16*)nullptr, BL, Dm, Dm, 0);
      // z = LN(z + gf)
      resln_kernel<<<BL / 8, 256, 0, stream>>>(zf, gf, ln1_s + (size_t)l * Dm,
                                               ln1_b + (size_t)l * Dm, zf, zh);
      // fused FFN -> gf
      ffn_kernel<<<BL / 64, 256, 0, stream>>>(zh, wff1_h + (size_t)l * FFd * Dm,
                                              b_ff1 + (size_t)l * FFd,
                                              wff2_h + (size_t)l * Dm * FFd,
                                              b_ff2 + (size_t)l * Dm, gf);
      // z = LN(z + gf)
      resln_kernel<<<BL / 8, 256, 0, stream>>>(zf, gf, ln2_s + (size_t)l * Dm,
                                               ln2_b + (size_t)l * Dm, zf, zh);
    }
    seqmean_kernel<<<Bsz, 128, 0, stream>>>(zf, hm_f, hm_h);
    gemm_kernel<<<dim3(Dm / 64, Bsz / 128), 256, 0, stream>>>(
        hm_h, w1_h, b1, h1_f, h1_h, Bsz, Dm, Dm, 2);
    gemm_kernel<<<dim3(Dm / 64, Bsz / 128), 256, 0, stream>>>(
        h1_h, w2_h, b2, branch_out, (_Float16*)nullptr, Bsz, Dm, Dm, 2);
  }

  l2norm_kernel<<<Bsz / 8, 256, 0, stream>>>(hb_f, out + NB_OFF);
  l2norm_kernel<<<Bsz / 8, 256, 0, stream>>>(out + HP_OFF, npn);
  l2norm_kernel<<<Bsz / 8, 256, 0, stream>>>(out + HN_OFF, nnn);
  logits_kernel<<<1, 256, 0, stream>>>(hb_f, w3, b3, out + LG_OFF);
  lse_kernel<<<Bsz, 128, 0, stream>>>(out + NB_OFF, npn, nnn, rowls);
  meanloss_kernel<<<1, 256, 0, stream>>>(rowls, out + LS_OFF);
}
